// SelfAttention_11321533792270
// MI455X (gfx1250) — compile-verified, matrix operand order checked by reading, not run on hardware
//
#include <hip/hip_runtime.h>
#include <math.h>

// ---------------------------------------------------------------------------
// Problem constants (reference: B=4, S=2048, F=1024, H=4, D=256)
// ---------------------------------------------------------------------------
#define B_  4
#define S_  2048
#define F_  1024
#define H_  4
#define D_  256
#define BS_ (B_ * S_)        // 8192 rows for the projection GEMMs

typedef __attribute__((ext_vector_type(16))) __bf16 v16bf;
typedef __attribute__((ext_vector_type(8)))  __bf16 v8bf;
typedef __attribute__((ext_vector_type(8)))  float  v8f;
typedef __attribute__((ext_vector_type(4)))  float  v4f;
typedef __attribute__((ext_vector_type(4)))  unsigned int v4u;
typedef __attribute__((ext_vector_type(8)))  int    v8i_;
typedef __attribute__((ext_vector_type(4)))  int    v4i_;

// ---------------------------------------------------------------------------
// WMMA helper: D = A(16x32 bf16) * B(32x16 bf16) + C(16x16 f32)
// ---------------------------------------------------------------------------
static __device__ __forceinline__ v8f wmma_bf16(v16bf a, v16bf b, v8f c) {
  return __builtin_amdgcn_wmma_f32_16x16x32_bf16(
      /*neg_a=*/false, a, /*neg_b=*/false, b,
      /*c_mod=*/(short)0, c, /*reuse_a=*/false, /*reuse_b=*/false);
}

// Fragment loader for a 16-row K-contiguous bf16 panel (row-major, ld).
//   lanes 0-15  hold row (lane)    with K = k0+0..7  and k0+16..23
//   lanes 16-31 hold row (lane-16) with K = k0+8..15 and k0+24..31
static __device__ __forceinline__ v16bf load_frag_bf16(const __bf16* panel,
                                                       int ld, int k0, int lane) {
  const int r  = lane & 15;
  const int kh = k0 + ((lane >> 4) << 3);
  const __bf16* p = panel + (size_t)r * ld + kh;
  v8bf lo = *reinterpret_cast<const v8bf*>(p);
  v8bf hi = *reinterpret_cast<const v8bf*>(p + 16);
  v16bf a;
#pragma unroll
  for (int i = 0; i < 8; ++i) { a[i] = lo[i]; a[i + 8] = hi[i]; }
  return a;
}

// Same fragment from an LDS-resident 32-wide panel (row stride = 32 elems).
static __device__ __forceinline__ v16bf load_frag_lds(const __bf16* panel,
                                                      int row0, int lane) {
  const int r  = row0 + (lane & 15);
  const int kh = (lane >> 4) << 3;
  const __bf16* p = panel + r * 32 + kh;
  v8bf lo = *reinterpret_cast<const v8bf*>(p);
  v8bf hi = *reinterpret_cast<const v8bf*>(p + 16);
  v16bf a;
#pragma unroll
  for (int i = 0; i < 8; ++i) { a[i] = lo[i]; a[i + 8] = hi[i]; }
  return a;
}

// f32 source (attention probabilities), converted to bf16 in-register.
static __device__ __forceinline__ v16bf load_frag_f32(const float* panel,
                                                      int ld, int k0, int lane) {
  const int r  = lane & 15;
  const int kh = k0 + ((lane >> 4) << 3);
  const float* p = panel + (size_t)r * ld + kh;
  v4f a0 = *reinterpret_cast<const v4f*>(p);
  v4f a1 = *reinterpret_cast<const v4f*>(p + 4);
  v4f b0 = *reinterpret_cast<const v4f*>(p + 16);
  v4f b1 = *reinterpret_cast<const v4f*>(p + 20);
  v16bf a;
#pragma unroll
  for (int i = 0; i < 4; ++i) {
    a[i]      = (__bf16)a0[i];
    a[i + 4]  = (__bf16)a1[i];
    a[i + 8]  = (__bf16)b0[i];
    a[i + 12] = (__bf16)b1[i];
  }
  return a;
}

static __device__ __forceinline__ int frag_row(int m_w, int mi, int lane, int e) {
  return m_w + mi * 16 + ((lane >> 4) << 3) + e;
}
static __device__ __forceinline__ int frag_col(int n_w, int ni, int lane) {
  return n_w + ni * 16 + (lane & 15);
}

// ---------------------------------------------------------------------------
// Tensor Data Mover: async DMA of a 2-D bf16 tile (tile_k x tile_rows,
// row stride = row_stride elems) from global memory into LDS.
// D# packing per CDNA5 ISA 8.3/8.4: group0 {count=1, lds_addr, global_addr,
// type=2}; group1 {data_size=2B, tensor_dim0/1, tile_dim0/1, dim0_stride}.
// This toolchain declares the 6-arg builtin; the extra int32x8 group is
// unused for 2-D tensors and passed zero-filled.
// ---------------------------------------------------------------------------
static __device__ __forceinline__ void tdm_load_tile_2d(unsigned lds_byte_addr,
                                                        const void* gaddr,
                                                        unsigned tile_k,
                                                        unsigned tile_rows,
                                                        unsigned row_stride) {
  const unsigned long long ga = (unsigned long long)gaddr;
  v4u g0;
  g0.x = 1u;                                            // count=1, user mode
  g0.y = lds_byte_addr;                                 // LDS dest (bytes)
  g0.z = (unsigned)ga;                                  // global_addr[31:0]
  g0.w = (unsigned)((ga >> 32) & 0x01FFFFFFu) | (2u << 30); // addr[56:32]|type=2
  v8i_ g1;
  g1[0] = (int)(1u << 16);                              // data_size = 2 bytes
  g1[1] = (int)((tile_k & 0xFFFFu) << 16);              // tensor_dim0[15:0]
  g1[2] = (int)(((tile_k >> 16) & 0xFFFFu) |            // tensor_dim0[31:16]
                ((tile_rows & 0xFFFFu) << 16));         // tensor_dim1[15:0]
  g1[3] = (int)(((tile_rows >> 16) & 0xFFFFu) |         // tensor_dim1[31:16]
                ((tile_k & 0xFFFFu) << 16));            // tile_dim0
  g1[4] = (int)(tile_rows & 0xFFFFu);                   // tile_dim1 (tile_dim2=0)
  g1[5] = (int)row_stride;                              // tensor_dim0_stride lo32
  g1[6] = 0;                                            // stride hi / dim1_stride
  g1[7] = 0;
  const v4i_ z4 = {0, 0, 0, 0};                         // 2-D: groups 2/3 unused
  const v8i_ z8 = {0, 0, 0, 0, 0, 0, 0, 0};             // extended group unused
  __builtin_amdgcn_tensor_load_to_lds(g0, g1, z4, z4, z8, 0);
}

static __device__ __forceinline__ unsigned lds_byte_offset(const void* p) {
  return (unsigned)(unsigned long long)
      (__attribute__((address_space(3))) const char*)p;
}

// ---------------------------------------------------------------------------
// Conversion kernels
// ---------------------------------------------------------------------------
__global__ __launch_bounds__(256) void cvt_f32_bf16(const float* __restrict__ in,
                                                    __bf16* __restrict__ out, int n) {
  int i = blockIdx.x * 256 + threadIdx.x;
  if (i < n) out[i] = (__bf16)in[i];
}

// Wt[n, k] = (bf16) W[k, n]   (F x F)
__global__ __launch_bounds__(256) void transpose_cvt(const float* __restrict__ W,
                                                     __bf16* __restrict__ Wt) {
  int i = blockIdx.x * 256 + threadIdx.x;      // i = k * F + n
  int k = i >> 10;
  int n = i & (F_ - 1);
  Wt[(size_t)n * F_ + k] = (__bf16)W[i];
}

// ---------------------------------------------------------------------------
// Projection GEMM (8192 x 1024 x 1024), weight panel TDM-staged through LDS.
//   A  : bf16 row-major activations (lda = F)
//   Bt : bf16 transposed weights (row n = column n of W, K-contiguous)
//   mode 0: dst = relu(acc + bias) -> [b,h,s,d]          (Q, K)
//   mode 1: dst = relu(acc + bias) -> [b,h,d,s]          (V transposed)
//   mode 2: outf = xres + (acc + bias) * gamma  (f32)    (output projection)
// Grid (F/128, BS/128), 256 threads = 8 waves (2x4), 64x32 tile per wave.
// ---------------------------------------------------------------------------
__global__ __launch_bounds__(256) void proj_gemm(const __bf16* __restrict__ A,
                                                 const __bf16* __restrict__ Bt,
                                                 const float* __restrict__ bias,
                                                 __bf16* __restrict__ dst,
                                                 const float* __restrict__ xres,
                                                 const float* __restrict__ gamma,
                                                 float* __restrict__ outf,
                                                 int mode) {
  __shared__ __bf16 smB[2][128 * 32];          // double-buffered weight panel
  const int lane = threadIdx.x & 31;
  const int wave = threadIdx.x >> 5;
  const int m_w  = blockIdx.y * 128 + (wave >> 2) * 64;
  const int n_w  = blockIdx.x * 128 + (wave & 3) * 32;
  const int n_l  = (wave & 3) * 32;            // wave's N offset inside panel

  const unsigned lds_base = lds_byte_offset(&smB[0][0]);
  const __bf16* gB = Bt + (size_t)(blockIdx.x * 128) * F_;

  v8f c[4][2] = {};
  for (int k0 = 0, it = 0; k0 < F_; k0 += 32, ++it) {
    const int buf = it & 1;
    if (threadIdx.x < 32) {                    // wave 0 drives the TDM
      tdm_load_tile_2d(lds_base + (unsigned)buf * (128 * 32 * 2),
                       gB + k0, 32u, 128u, (unsigned)F_);
      __builtin_amdgcn_s_wait_tensorcnt(0);
    }
    __syncthreads();                           // publish panel to all waves

    v16bf af[4], bf[2];
#pragma unroll
    for (int mi = 0; mi < 4; ++mi) {
      const __bf16* pa = A + (size_t)(m_w + mi * 16) * F_;
      af[mi] = load_frag_bf16(pa, F_, k0, lane);
      __builtin_prefetch(pa + k0 + 32, 0, 1);  // speculative, unconditional
    }
#pragma unroll
    for (int ni = 0; ni < 2; ++ni)
      bf[ni] = load_frag_lds(&smB[buf][0], n_l + ni * 16, lane);
#pragma unroll
    for (int mi = 0; mi < 4; ++mi)
#pragma unroll
      for (int ni = 0; ni < 2; ++ni)
        c[mi][ni] = wmma_bf16(af[mi], bf[ni], c[mi][ni]);
  }

  if (mode == 2) {                             // output projection epilogue
    const float g = gamma[0];
#pragma unroll
    for (int mi = 0; mi < 4; ++mi)
#pragma unroll
      for (int ni = 0; ni < 2; ++ni) {
        const int col = frag_col(n_w, ni, lane);
        const float bv = bias[col];
#pragma unroll
        for (int e = 0; e < 8; ++e) {
          const int row = frag_row(m_w, mi, lane, e);
          const size_t idx = (size_t)row * F_ + col;
          outf[idx] = xres[idx] + (c[mi][ni][e] + bv) * g;
        }
      }
  } else {                                     // QKV epilogue: bias + ReLU
#pragma unroll
    for (int mi = 0; mi < 4; ++mi)
#pragma unroll
      for (int ni = 0; ni < 2; ++ni) {
        const int col = frag_col(n_w, ni, lane);
        const int h   = col >> 8;              // col / D
        const int dd  = col & (D_ - 1);
        const float bv = bias[col];
#pragma unroll
        for (int e = 0; e < 8; ++e) {
          const int row = frag_row(m_w, mi, lane, e);
          const int bb  = row >> 11;           // row / S
          const int s   = row & (S_ - 1);
          float v = fmaxf(c[mi][ni][e] + bv, 0.0f);
          size_t idx;
          if (mode == 0)
            idx = ((size_t)((bb * H_ + h) * S_ + s)) * D_ + dd;
          else
            idx = ((size_t)((bb * H_ + h) * D_ + dd)) * S_ + s;
          dst[idx] = (__bf16)v;
        }
      }
  }
}

// ---------------------------------------------------------------------------
// Logits: attn[bh, s, t] = (Q[bh,s,:] . K[bh,t,:]) / sqrt(D)    (f32 out)
// Grid: (S/128, S/128, B*H)
// ---------------------------------------------------------------------------
__global__ __launch_bounds__(256) void logits_gemm(const __bf16* __restrict__ Q,
                                                   const __bf16* __restrict__ Kb,
                                                   float* __restrict__ attn) {
  const int lane = threadIdx.x & 31;
  const int wave = threadIdx.x >> 5;
  const int bh   = blockIdx.z;
  const int m_w  = blockIdx.y * 128 + (wave >> 2) * 64;
  const int n_w  = blockIdx.x * 128 + (wave & 3) * 32;

  const __bf16* Aq = Q  + (size_t)bh * S_ * D_;
  const __bf16* Bk = Kb + (size_t)bh * S_ * D_;
  float* out = attn + (size_t)bh * S_ * S_;

  v8f c[4][2] = {};
  for (int k0 = 0; k0 < D_; k0 += 32) {
    v16bf af[4], bf[2];
#pragma unroll
    for (int mi = 0; mi < 4; ++mi)
      af[mi] = load_frag_bf16(Aq + (size_t)(m_w + mi * 16) * D_, D_, k0, lane);
#pragma unroll
    for (int ni = 0; ni < 2; ++ni)
      bf[ni] = load_frag_bf16(Bk + (size_t)(n_w + ni * 16) * D_, D_, k0, lane);
#pragma unroll
    for (int mi = 0; mi < 4; ++mi)
#pragma unroll
      for (int ni = 0; ni < 2; ++ni)
        c[mi][ni] = wmma_bf16(af[mi], bf[ni], c[mi][ni]);
  }

  const float scale = 0.0625f;            // 1/sqrt(256)
#pragma unroll
  for (int mi = 0; mi < 4; ++mi)
#pragma unroll
    for (int ni = 0; ni < 2; ++ni) {
      const int col = frag_col(n_w, ni, lane);
#pragma unroll
      for (int e = 0; e < 8; ++e) {
        const int row = frag_row(m_w, mi, lane, e);
        out[(size_t)row * S_ + col] = c[mi][ni][e] * scale;
      }
    }
}

// ---------------------------------------------------------------------------
// Row softmax over attn rows of length S (in place, f32).
// ---------------------------------------------------------------------------
__global__ __launch_bounds__(256) void softmax_rows(float* __restrict__ attn) {
  __shared__ float sm[256];
  const int tid = threadIdx.x;
  float* p = attn + (size_t)blockIdx.x * S_;

  float v[8];
#pragma unroll
  for (int i = 0; i < 8; ++i) v[i] = p[tid + i * 256];

  float mx = v[0];
#pragma unroll
  for (int i = 1; i < 8; ++i) mx = fmaxf(mx, v[i]);
  sm[tid] = mx;
  __syncthreads();
  for (int off = 128; off > 0; off >>= 1) {
    if (tid < off) sm[tid] = fmaxf(sm[tid], sm[tid + off]);
    __syncthreads();
  }
  const float rmax = sm[0];
  __syncthreads();

  float sum = 0.0f;
#pragma unroll
  for (int i = 0; i < 8; ++i) { v[i] = __expf(v[i] - rmax); sum += v[i]; }
  sm[tid] = sum;
  __syncthreads();
  for (int off = 128; off > 0; off >>= 1) {
    if (tid < off) sm[tid] += sm[tid + off];
    __syncthreads();
  }
  const float inv = 1.0f / sm[0];
#pragma unroll
  for (int i = 0; i < 8; ++i) p[tid + i * 256] = v[i] * inv;
}

// ---------------------------------------------------------------------------
// ctx = P * V :  concat[b, s, h*D+d] = sum_t attn[bh,s,t] * V[bh,t,d]
// Grid: (D/128, S/128, B*H)
// ---------------------------------------------------------------------------
__global__ __launch_bounds__(256) void ctx_gemm(const float* __restrict__ attn,
                                                const __bf16* __restrict__ Vt,
                                                __bf16* __restrict__ concat) {
  const int lane = threadIdx.x & 31;
  const int wave = threadIdx.x >> 5;
  const int bh   = blockIdx.z;
  const int m_w  = blockIdx.y * 128 + (wave >> 2) * 64;
  const int n_w  = blockIdx.x * 128 + (wave & 3) * 32;

  const float*  Ap = attn + (size_t)bh * S_ * S_;
  const __bf16* Bp = Vt   + (size_t)bh * D_ * S_;

  v8f c[4][2] = {};
  for (int k0 = 0; k0 < S_; k0 += 32) {
    v16bf af[4], bf[2];
#pragma unroll
    for (int mi = 0; mi < 4; ++mi) {
      const float* pa = Ap + (size_t)(m_w + mi * 16) * S_;
      af[mi] = load_frag_f32(pa, S_, k0, lane);
      __builtin_prefetch(pa + k0 + 32, 0, 1);
    }
#pragma unroll
    for (int ni = 0; ni < 2; ++ni)
      bf[ni] = load_frag_bf16(Bp + (size_t)(n_w + ni * 16) * S_, S_, k0, lane);
#pragma unroll
    for (int mi = 0; mi < 4; ++mi)
#pragma unroll
      for (int ni = 0; ni < 2; ++ni)
        c[mi][ni] = wmma_bf16(af[mi], bf[ni], c[mi][ni]);
  }

  const int bb = bh >> 2;                 // bh / H
  const int h  = bh & (H_ - 1);
#pragma unroll
  for (int mi = 0; mi < 4; ++mi)
#pragma unroll
    for (int ni = 0; ni < 2; ++ni) {
      const int dd = frag_col(n_w, ni, lane);
#pragma unroll
      for (int e = 0; e < 8; ++e) {
        const int s = frag_row(m_w, mi, lane, e);
        concat[((size_t)(bb * S_ + s)) * F_ + h * D_ + dd] = (__bf16)c[mi][ni][e];
      }
    }
}

// ---------------------------------------------------------------------------
// Host-side orchestration
// ---------------------------------------------------------------------------
extern "C" void kernel_launch(void* const* d_in, const int* in_sizes, int n_in,
                              void* d_out, int out_size, void* d_ws, size_t ws_size,
                              hipStream_t stream) {
  (void)in_sizes; (void)n_in; (void)out_size; (void)ws_size;

  const float* x     = (const float*)d_in[0];
  const float* Wq    = (const float*)d_in[1];
  const float* bq    = (const float*)d_in[2];
  const float* Wk    = (const float*)d_in[3];
  const float* bk    = (const float*)d_in[4];
  const float* Wv    = (const float*)d_in[5];
  const float* bv    = (const float*)d_in[6];
  const float* Wo    = (const float*)d_in[7];
  const float* bo    = (const float*)d_in[8];
  const float* gamma = (const float*)d_in[9];

  float* out_ptr  = (float*)d_out;                       // [B,S,F]
  float* attn_ptr = (float*)d_out + (size_t)BS_ * F_;    // [B,H,S,S]

  // Workspace layout (bf16 buffers), ~88 MB total.
  char* w = (char*)d_ws;
  const size_t XB  = (size_t)BS_ * F_ * 2;   // 16 MB
  const size_t WT  = (size_t)F_ * F_ * 2;    // 2 MB
  __bf16* xb    = (__bf16*)(w);
  __bf16* wq_t  = (__bf16*)(w + XB);
  __bf16* wk_t  = (__bf16*)(w + XB + WT);
  __bf16* wv_t  = (__bf16*)(w + XB + 2 * WT);
  __bf16* wo_t  = (__bf16*)(w + XB + 3 * WT);
  __bf16* qbuf  = (__bf16*)(w + XB + 4 * WT);
  __bf16* kbuf  = (__bf16*)(w + 2 * XB + 4 * WT);
  __bf16* vtbuf = (__bf16*)(w + 3 * XB + 4 * WT);
  __bf16* ccbuf = (__bf16*)(w + 4 * XB + 4 * WT);

  // 1) Precision conversion + weight transposes.
  cvt_f32_bf16<<<(BS_ * F_) / 256, 256, 0, stream>>>(x, xb, BS_ * F_);
  transpose_cvt<<<(F_ * F_) / 256, 256, 0, stream>>>(Wq, wq_t);
  transpose_cvt<<<(F_ * F_) / 256, 256, 0, stream>>>(Wk, wk_t);
  transpose_cvt<<<(F_ * F_) / 256, 256, 0, stream>>>(Wv, wv_t);
  transpose_cvt<<<(F_ * F_) / 256, 256, 0, stream>>>(Wo, wo_t);

  // 2) Q/K/V projections (WMMA + TDM-staged weights).
  dim3 gproj(F_ / 128, BS_ / 128);
  proj_gemm<<<gproj, 256, 0, stream>>>(xb, wq_t, bq, qbuf,  nullptr, nullptr, nullptr, 0);
  proj_gemm<<<gproj, 256, 0, stream>>>(xb, wk_t, bk, kbuf,  nullptr, nullptr, nullptr, 0);
  proj_gemm<<<gproj, 256, 0, stream>>>(xb, wv_t, bv, vtbuf, nullptr, nullptr, nullptr, 1);

  // 3) Attention logits -> d_out attn region (f32, scaled).
  dim3 glog(S_ / 128, S_ / 128, B_ * H_);
  logits_gemm<<<glog, 256, 0, stream>>>(qbuf, kbuf, attn_ptr);

  // 4) Row softmax in place.
  softmax_rows<<<B_ * H_ * S_, 256, 0, stream>>>(attn_ptr);

  // 5) ctx = P * V  -> merged-head bf16 buffer.
  dim3 gctx(D_ / 128, S_ / 128, B_ * H_);
  ctx_gemm<<<gctx, 256, 0, stream>>>(attn_ptr, vtbuf, ccbuf);

  // 6) Output projection + residual + gamma (TDM-staged Wo).
  proj_gemm<<<gproj, 256, 0, stream>>>(ccbuf, wo_t, bo, nullptr, x, gamma, out_ptr, 2);
}
